// LeWinTransformerBlock_20942260535762
// MI455X (gfx1250) — matrix operand order repaired and checked by reading, not compiled, not run on hardware
//
#include <hip/hip_runtime.h>
#include <hip/hip_bf16.h>
#include <math.h>

typedef __attribute__((ext_vector_type(16))) _Float16 v16h;
typedef __attribute__((ext_vector_type(8)))  _Float16 v8h;
typedef __attribute__((ext_vector_type(8)))  float    v8f;
typedef __attribute__((ext_vector_type(4)))  int      v4i;

#define DEVI __device__ __forceinline__

// CDNA5 async global->LDS copy path (ASYNCcnt), if the toolchain exposes it.
#if __has_builtin(__builtin_amdgcn_global_load_async_to_lds_b128) && \
    __has_builtin(__builtin_amdgcn_s_wait_asynccnt)
#define HAVE_ASYNC_LDS 1
typedef __attribute__((address_space(1))) v4i as1_v4i;  // global int4
typedef __attribute__((address_space(3))) v4i as3_v4i;  // LDS int4
#endif

// ---------------------------------------------------------------------------
// WMMA fragment loaders (CDNA5 16-bit layouts, wave32)
// A: 16x32 f16 row-major.  lane<16: K=k0+{0..7,16..23}, row=lane
//                          lane>=16: K=k0+{8..15,24..31}, row=lane-16
// ---------------------------------------------------------------------------
DEVI v16h load_a16(const _Float16* A, int lda, int m, int k0, int lane) {
    const _Float16* p = A + (size_t)m * lda + k0 + ((lane >> 4) << 3);
    v8h lo = *(const v8h*)(p);
    v8h hi = *(const v8h*)(p + 16);
    v16h r;
#pragma unroll
    for (int i = 0; i < 8; ++i) { r[i] = lo[i]; r[i + 8] = hi[i]; }
    return r;
}

// B: 32x16 f16 from N-major ("B^T") source BT[N][K].
DEVI v16h load_bt16(const _Float16* BT, int ldb, int n, int k0, int lane) {
    const _Float16* p = BT + (size_t)n * ldb + k0 + ((lane >> 4) << 3);
    v8h lo = *(const v8h*)(p);
    v8h hi = *(const v8h*)(p + 16);
    v16h r;
#pragma unroll
    for (int i = 0; i < 8; ++i) { r[i] = lo[i]; r[i + 8] = hi[i]; }
    return r;
}

DEVI v8f wmma16(v16h a, v16h b, v8f c) {
    return __builtin_amdgcn_wmma_f32_16x16x32_f16(false, a, false, b, (short)0, c, false, false);
}

DEVI float gelu_f(float x) { return 0.5f * x * (1.0f + erff(x * 0.70710678118654752f)); }

// ---------------------------------------------------------------------------
// Weight prep: fp32 row-major [K,N] -> f16 N-major [N,K] (lives in 192MB L2)
// ---------------------------------------------------------------------------
__global__ void cvt_tr(const float* __restrict__ W, _Float16* __restrict__ WT, int K, int N) {
    int i = blockIdx.x * blockDim.x + threadIdx.x;
    if (i >= K * N) return;
    int n = i % N, k = i / N;
    WT[(size_t)n * K + k] = (_Float16)W[i];
}

// ---------------------------------------------------------------------------
// LayerNorm (C=256) -> f16; wave-per-token; optional window-partition reorder
// ---------------------------------------------------------------------------
__global__ __launch_bounds__(256) void ln_f16(const float* __restrict__ x,
                                              const float* __restrict__ g,
                                              const float* __restrict__ bta,
                                              _Float16* __restrict__ out,
                                              int window_order) {
    int lane = threadIdx.x & 31, wave = threadIdx.x >> 5;
    size_t tok = (size_t)blockIdx.x * 8 + wave;
    const float* px = x + tok * 256 + lane * 8;
    float v[8];
#pragma unroll
    for (int i = 0; i < 8; ++i) v[i] = px[i];
    float s = 0.f;
#pragma unroll
    for (int i = 0; i < 8; ++i) s += v[i];
#pragma unroll
    for (int off = 16; off > 0; off >>= 1) s += __shfl_xor(s, off, 32);
    float mean = s * (1.0f / 256.0f);
    float q = 0.f;
#pragma unroll
    for (int i = 0; i < 8; ++i) { float d = v[i] - mean; q += d * d; }
#pragma unroll
    for (int off = 16; off > 0; off >>= 1) q += __shfl_xor(q, off, 32);
    float rinv = rsqrtf(q * (1.0f / 256.0f) + 1e-5f);

    size_t drow = tok;
    if (window_order) {
        int b = (int)(tok >> 14);
        int l = (int)(tok & 16383);
        int hI = l >> 7, wI = l & 127;
        int win = ((b * 16 + (hI >> 3)) * 16 + (wI >> 3));
        int rr = (hI & 7) * 8 + (wI & 7);
        drow = (size_t)win * 64 + rr;
    }
    _Float16* po = out + drow * 256 + lane * 8;
#pragma unroll
    for (int i = 0; i < 8; ++i) {
        int c = lane * 8 + i;
        po[i] = (_Float16)((v[i] - mean) * rinv * g[c] + bta[c]);
    }
}

// ---------------------------------------------------------------------------
// Fully fused windowed attention: one block per window (2048 blocks, 8 waves).
// QKV projections, per-head S=QK^T+bias, softmax, PV, output proj + residual —
// all intermediates live in LDS (no Q/K/V/attn-out HBM round-trips).
// ---------------------------------------------------------------------------
__global__ __launch_bounds__(256) void attn_fused(
    const _Float16* __restrict__ xn,      // [2048][64][256] window-ordered LN1 out
    const _Float16* __restrict__ wqT,     // [256][256]
    const float* __restrict__ bq,
    const _Float16* __restrict__ wkvT,    // [512][256]
    const float* __restrict__ bkv,
    const float* __restrict__ rpb,        // [225][8]
    const _Float16* __restrict__ wprojT,  // [256][256]
    const float* __restrict__ bproj,
    const float* __restrict__ xres,       // x (token order, residual)
    float* __restrict__ x1) {
    __shared__ _Float16 Qs[8][64][32];    // 32 KB
    __shared__ _Float16 Ks[8][64][32];    // 32 KB
    __shared__ _Float16 Vs[8][32][64];    // 32 KB (V^T per head)
    __shared__ float    S[64][65];        // 16.25 KB
    __shared__ _Float16 P[64][72];        // 9 KB
    __shared__ _Float16 Of[64][256];      // 32 KB

    int lane = threadIdx.x & 31, wave = threadIdx.x >> 5;
    int win = blockIdx.x;
    const _Float16* Xw = xn + (size_t)win * 64 * 256;
    const float qscale = 0.17677669529663687f;

    // ---- Q = Xw @ wqT + bq (scaled) -> Qs[head][n][d] (64 tiles, 8 per wave)
    for (int t = wave * 8; t < wave * 8 + 8; ++t) {
        int tm = t & 3, tn = t >> 2;
        v8f acc = {};
        for (int k = 0; k < 256; k += 32) {
            v16h a = load_a16(Xw, 256, tm * 16 + (lane & 15), k, lane);
            v16h b = load_bt16(wqT, 256, tn * 16 + (lane & 15), k, lane);
            acc = wmma16(a, b, acc);
        }
        int col = tn * 16 + (lane & 15);
        int hh = col >> 5, d = col & 31;
        int rb = tm * 16 + ((lane >> 4) << 3);
        float bv = bq[col];
#pragma unroll
        for (int r = 0; r < 8; ++r)
            Qs[hh][rb + r][d] = (_Float16)((acc[r] + bv) * qscale);
    }
    // ---- K,V = Xw @ wkvT + bkv -> Ks[h][n][d], Vs[h][d][n] (128 tiles, 16/wave)
    for (int t = wave * 16; t < wave * 16 + 16; ++t) {
        int tm = t & 3, tn = t >> 2;
        v8f acc = {};
        for (int k = 0; k < 256; k += 32) {
            v16h a = load_a16(Xw, 256, tm * 16 + (lane & 15), k, lane);
            v16h b = load_bt16(wkvT, 256, tn * 16 + (lane & 15), k, lane);
            acc = wmma16(a, b, acc);
        }
        int col = tn * 16 + (lane & 15);
        int rb = tm * 16 + ((lane >> 4) << 3);
        float bv = bkv[col];
        if (col < 256) {
            int hh = col >> 5, d = col & 31;
#pragma unroll
            for (int r = 0; r < 8; ++r) Ks[hh][rb + r][d] = (_Float16)(acc[r] + bv);
        } else {
            int c2 = col - 256, hh = c2 >> 5, d = c2 & 31;
#pragma unroll
            for (int r = 0; r < 8; ++r) Vs[hh][d][rb + r] = (_Float16)(acc[r] + bv);
        }
    }
    __syncthreads();

    // ---- per head: S = Q@K^T + relpos bias; softmax; O = P@V
    for (int h = 0; h < 8; ++h) {
        for (int t = wave; t < 16; t += 8) {
            int tm = t >> 2, tn = t & 3;
            v16h a = load_a16(&Qs[h][0][0], 32, tm * 16 + (lane & 15), 0, lane);
            v16h b = load_bt16(&Ks[h][0][0], 32, tn * 16 + (lane & 15), 0, lane);
            v8f s = {};
            s = wmma16(a, b, s);
            int col = tn * 16 + (lane & 15);
            int rb = tm * 16 + ((lane >> 4) << 3);
#pragma unroll
            for (int r = 0; r < 8; ++r) {
                int row = rb + r;
                int dy = (row >> 3) - (col >> 3) + 7;
                int dx = (row & 7) - (col & 7) + 7;
                S[row][col] = s[r] + rpb[(dy * 15 + dx) * 8 + h];
            }
        }
        __syncthreads();
#pragma unroll
        for (int i = 0; i < 8; ++i) {
            int row = wave * 8 + i;
            float v0 = S[row][lane], v1 = S[row][lane + 32];
            float mx = fmaxf(v0, v1);
#pragma unroll
            for (int off = 16; off > 0; off >>= 1) mx = fmaxf(mx, __shfl_xor(mx, off, 32));
            float e0 = expf(v0 - mx), e1 = expf(v1 - mx);
            float sm = e0 + e1;
#pragma unroll
            for (int off = 16; off > 0; off >>= 1) sm += __shfl_xor(sm, off, 32);
            float inv = 1.0f / sm;
            P[row][lane]      = (_Float16)(e0 * inv);
            P[row][lane + 32] = (_Float16)(e1 * inv);
        }
        __syncthreads();
        {
            int tm = wave >> 1, tn = wave & 1;
            v8f o = {};
#pragma unroll
            for (int kk = 0; kk < 64; kk += 32) {
                v16h a = load_a16(&P[0][0], 72, tm * 16 + (lane & 15), kk, lane);
                v16h b = load_bt16(&Vs[h][0][0], 64, tn * 16 + (lane & 15), kk, lane);
                o = wmma16(a, b, o);
            }
            int col = tn * 16 + (lane & 15);
            int rb = tm * 16 + ((lane >> 4) << 3);
#pragma unroll
            for (int r = 0; r < 8; ++r) Of[rb + r][h * 32 + col] = (_Float16)o[r];
        }
        __syncthreads();
    }

    // ---- proj + residual (window -> token reorder), f32 out
    int bb = win >> 8, wh = (win >> 4) & 15, ww = win & 15;
    for (int t = wave * 8; t < wave * 8 + 8; ++t) {
        int tm = t & 3, tn = t >> 2;
        v8f acc = {};
        for (int k = 0; k < 256; k += 32) {
            v16h a = load_a16(&Of[0][0], 256, tm * 16 + (lane & 15), k, lane);
            v16h b = load_bt16(wprojT, 256, tn * 16 + (lane & 15), k, lane);
            acc = wmma16(a, b, acc);
        }
        int col = tn * 16 + (lane & 15);
        int rb = tm * 16 + ((lane >> 4) << 3);
        float bv = bproj[col];
#pragma unroll
        for (int r = 0; r < 8; ++r) {
            int rr = rb + r;
            int ih = rr >> 3, iw = rr & 7;
            size_t tok = ((size_t)bb * 128 + wh * 8 + ih) * 128 + (ww * 8 + iw);
            x1[tok * 256 + col] = xres[tok * 256 + col] + acc[r] + bv;
        }
    }
}

// ---------------------------------------------------------------------------
// w1 GEMM + exact GELU -> f16 hidden.  Wave: 16x64 strip, block: 128x64.
// ---------------------------------------------------------------------------
__global__ __launch_bounds__(256) void gemm_gelu(const _Float16* __restrict__ A,
                                                 const _Float16* __restrict__ BT,
                                                 const float* __restrict__ bias,
                                                 _Float16* __restrict__ out,
                                                 int N, int K) {
    int lane = threadIdx.x & 31, wave = threadIdx.x >> 5;
    int row0 = blockIdx.y * 128 + wave * 16;
    int n0   = blockIdx.x * 64;
    v8f acc[4] = {};
    int m = row0 + (lane & 15);
    for (int k = 0; k < K; k += 32) {
        if (k + 32 < K) __builtin_prefetch(A + (size_t)m * K + k + 32, 0, 1);
        v16h a = load_a16(A, K, m, k, lane);
#pragma unroll
        for (int j = 0; j < 4; ++j) {
            v16h b = load_bt16(BT, K, n0 + j * 16 + (lane & 15), k, lane);
            acc[j] = wmma16(a, b, acc[j]);
        }
    }
    int rbase = row0 + ((lane >> 4) << 3);
#pragma unroll
    for (int j = 0; j < 4; ++j) {
        int col = n0 + j * 16 + (lane & 15);
        float bv = bias[col];
#pragma unroll
        for (int r = 0; r < 8; ++r)
            out[(size_t)(rbase + r) * N + col] = (_Float16)gelu_f(acc[j][r] + bv);
    }
}

// ---------------------------------------------------------------------------
// Fused depthwise 3x3 + GELU + w2 GEMM + residual -> f32 output.
// Block = 8x16 spatial token tile (one batch image), all 256 output columns.
// Per 32-channel K-step: stage 10x18 halo patch of y1 into LDS (async
// global->LDS when available), conv+GELU into LDS A-tile, WMMA-accumulate
// 16 col-tiles per wave.  y2 never touches HBM.
// ---------------------------------------------------------------------------
__global__ __launch_bounds__(256) void dwconv_w2_fused(
    const _Float16* __restrict__ y1,   // [B][128][128][1024]
    const float* __restrict__ wk,      // [1024][9]
    const float* __restrict__ wb,      // [1024]
    const _Float16* __restrict__ w2T,  // [256][1024]
    const float* __restrict__ b2,
    const float* __restrict__ x1,      // residual (token order)
    float* __restrict__ out) {
    __shared__ _Float16 patch[10][18][32];  // 11.25 KB
    __shared__ _Float16 atile[128][32];     // 8 KB
    __shared__ float    wks[32][10];        // conv taps for this K-step
    __shared__ float    wbs[32];

    int tid = threadIdx.x;
    int lane = tid & 31, wave = tid >> 5;
    int s = blockIdx.x;                 // spatial tile 0..1023
    int b = s >> 7;
    int h0 = ((s >> 3) & 15) * 8;
    int w0 = (s & 7) * 16;

    v8f acc[16] = {};
    const _Float16 hz = (_Float16)0.f;
    const v8h zv = {hz, hz, hz, hz, hz, hz, hz, hz};

    for (int kk = 0; kk < 1024; kk += 32) {
        // stage halo patch (zero-padded) + conv taps
        for (int ppos = tid; ppos < 180; ppos += 256) {
            int ph = ppos / 18, pw = ppos % 18;
            int hh = h0 + ph - 1, ww2 = w0 + pw - 1;
            _Float16* dstp = &patch[ph][pw][0];
            if (hh >= 0 && hh < 128 && ww2 >= 0 && ww2 < 128) {
                const _Float16* src = y1 + (((size_t)b * 128 + hh) * 128 + ww2) * 1024 + kk;
#ifdef HAVE_ASYNC_LDS
                // direct global->LDS DMA, tracked by ASYNCcnt (no VGPR staging)
                __builtin_amdgcn_global_load_async_to_lds_b128(
                    (as1_v4i*)(src),      (as3_v4i*)(dstp),      0, 0);
                __builtin_amdgcn_global_load_async_to_lds_b128(
                    (as1_v4i*)(src + 8),  (as3_v4i*)(dstp + 8),  0, 0);
                __builtin_amdgcn_global_load_async_to_lds_b128(
                    (as1_v4i*)(src + 16), (as3_v4i*)(dstp + 16), 0, 0);
                __builtin_amdgcn_global_load_async_to_lds_b128(
                    (as1_v4i*)(src + 24), (as3_v4i*)(dstp + 24), 0, 0);
#else
                v8h* dst = (v8h*)dstp;
                const v8h* sp = (const v8h*)src;
                dst[0] = sp[0]; dst[1] = sp[1]; dst[2] = sp[2]; dst[3] = sp[3];
#endif
            } else {
                v8h* dst = (v8h*)dstp;
                dst[0] = zv; dst[1] = zv; dst[2] = zv; dst[3] = zv;
            }
        }
        for (int i = tid; i < 288; i += 256) wks[i / 9][i % 9] = wk[(size_t)(kk + i / 9) * 9 + i % 9];
        if (tid < 32) wbs[tid] = wb[kk + tid];
#ifdef HAVE_ASYNC_LDS
        __builtin_amdgcn_s_wait_asynccnt(0);
#endif
        __syncthreads();

        // depthwise conv + GELU -> A-tile [128 tokens][32 ch]
        {
            int r = tid >> 1;
            int cb = (tid & 1) << 4;
            int ih = r >> 4, iw = r & 15;
#pragma unroll
            for (int e = 0; e < 16; ++e) {
                int c = cb + e;
                float a = wbs[c];
#pragma unroll
                for (int dy = 0; dy < 3; ++dy)
#pragma unroll
                    for (int dx = 0; dx < 3; ++dx)
                        a += (float)patch[ih + dy][iw + dx][c] * wks[c][dy * 3 + dx];
                atile[r][c] = (_Float16)gelu_f(a);
            }
        }
        __syncthreads();

        // WMMA: wave owns 16 rows x 256 cols, accumulate across kk
        v16h a = load_a16(&atile[0][0], 32, wave * 16 + (lane & 15), 0, lane);
#pragma unroll
        for (int j = 0; j < 16; ++j) {
            v16h bf = load_bt16(w2T, 1024, j * 16 + (lane & 15), kk, lane);
            acc[j] = wmma16(a, bf, acc[j]);
        }
        __syncthreads();
    }

    // epilogue: + b2 + residual -> f32 out
    int rbase = wave * 16 + ((lane >> 4) << 3);
#pragma unroll
    for (int j = 0; j < 16; ++j) {
        int col = j * 16 + (lane & 15);
        float bv = b2[col];
#pragma unroll
        for (int r = 0; r < 8; ++r) {
            int rr = rbase + r;
            size_t tok = ((size_t)b * 128 + h0 + (rr >> 4)) * 128 + (w0 + (rr & 15));
            out[tok * 256 + col] = x1[tok * 256 + col] + acc[j][r] + bv;
        }
    }
}

// ---------------------------------------------------------------------------
// Host launch
// ---------------------------------------------------------------------------
extern "C" void kernel_launch(void* const* d_in, const int* in_sizes, int n_in,
                              void* d_out, int out_size, void* d_ws, size_t ws_size,
                              hipStream_t stream) {
    const float* x     = (const float*)d_in[0];
    const float* n1g   = (const float*)d_in[1];
    const float* n1b   = (const float*)d_in[2];
    const float* wq    = (const float*)d_in[3];
    const float* bq    = (const float*)d_in[4];
    const float* wkv   = (const float*)d_in[5];
    const float* bkv   = (const float*)d_in[6];
    const float* rpb   = (const float*)d_in[7];
    const float* wproj = (const float*)d_in[8];
    const float* bproj = (const float*)d_in[9];
    const float* n2g   = (const float*)d_in[10];
    const float* n2b   = (const float*)d_in[11];
    const float* w1    = (const float*)d_in[12];
    const float* b1    = (const float*)d_in[13];
    const float* dwk   = (const float*)d_in[14];
    const float* dwb   = (const float*)d_in[15];
    const float* w2    = (const float*)d_in[16];
    const float* b2    = (const float*)d_in[17];
    float* out = (float*)d_out;

    const int M = 131072;  // B*L tokens (= 2048 windows * 64)

    char* p = (char*)d_ws;
    auto alloc_h = [&](size_t elems) { _Float16* r = (_Float16*)p; p += elems * 2; return r; };
    _Float16* wqT    = alloc_h((size_t)256 * 256);
    _Float16* wkvT   = alloc_h((size_t)512 * 256);
    _Float16* wprojT = alloc_h((size_t)256 * 256);
    _Float16* w1T    = alloc_h((size_t)1024 * 256);
    _Float16* w2T    = alloc_h((size_t)256 * 1024);
    _Float16* xn = alloc_h((size_t)M * 256);          // LN1 (window order), reused for LN2
    float* x1 = (float*)p; p += (size_t)M * 256 * 4;  // x + attention output (residual)
    _Float16* y1 = alloc_h((size_t)M * 1024);         // gelu(w1 h2 + b1)

    // 1. weights -> f16 transposed (resident in L2 thereafter)
    cvt_tr<<<(256 * 256 + 255) / 256, 256, 0, stream>>>(wq, wqT, 256, 256);
    cvt_tr<<<(256 * 512 + 255) / 256, 256, 0, stream>>>(wkv, wkvT, 256, 512);
    cvt_tr<<<(256 * 256 + 255) / 256, 256, 0, stream>>>(wproj, wprojT, 256, 256);
    cvt_tr<<<(256 * 1024 + 255) / 256, 256, 0, stream>>>(w1, w1T, 256, 1024);
    cvt_tr<<<(1024 * 256 + 255) / 256, 256, 0, stream>>>(w2, w2T, 1024, 256);

    // 2. LN1 + window partition -> f16
    ln_f16<<<M / 8, 256, 0, stream>>>(x, n1g, n1b, xn, 1);

    // 3. fully fused window attention (QKV + softmax + PV + proj + residual)
    attn_fused<<<2048, 256, 0, stream>>>(xn, wqT, bq, wkvT, bkv, rpb, wprojT, bproj, x, x1);

    // 4. LN2 -> f16 (token order)
    ln_f16<<<M / 8, 256, 0, stream>>>(x1, n2g, n2b, xn, 0);

    // 5. w1 + GELU -> f16 hidden
    gemm_gelu<<<dim3(16, M / 128), 256, 0, stream>>>(xn, w1T, b1, y1, 1024, 256);

    // 6. fused depthwise 3x3 + GELU + w2 + residual -> f32 output
    dwconv_w2_fused<<<1024, 256, 0, stream>>>(y1, dwk, dwb, w2T, b2, x1, out);
}
